// Transformer_58514634441358
// MI455X (gfx1250) — compile-verified, matrix operand order checked by reading, not run on hardware
//
#include <hip/hip_runtime.h>
#include <stdint.h>

#define D_MODEL 512
#define NHEAD 8
#define DHEAD 64
#define SEQ_T 256
#define BATCH 16
#define NROWS (BATCH*SEQ_T)   // 4096
#define FF_DIM 2048
#define VOCAB 8000
#define NEG_BIG (-4294967296.0f)

typedef __bf16 v16bf __attribute__((ext_vector_type(16)));
typedef float  v8f   __attribute__((ext_vector_type(8)));
typedef int    v4i   __attribute__((vector_size(16)));   // matches builtin param type

union FragBF { v16bf v; unsigned int u[8]; };

__device__ __forceinline__ unsigned short f2bf(float f){
  unsigned int u = __float_as_uint(f);
  u += 0x7FFFu + ((u >> 16) & 1u);          // round-to-nearest-even
  return (unsigned short)(u >> 16);
}
__device__ __forceinline__ unsigned int packbf(float a, float b){
  return (unsigned int)f2bf(a) | ((unsigned int)f2bf(b) << 16);
}

__device__ __forceinline__ v8f wmma_bf16(v16bf a, v16bf b, v8f c){
  return __builtin_amdgcn_wmma_f32_16x16x32_bf16(false, a, false, b, (short)0, c, false, false);
}

// ----- async 16B global->LDS copy (CDNA5 GLOBAL_LOAD_ASYNC_TO_LDS_B128) -----
#if defined(__HIP_DEVICE_COMPILE__) && __has_builtin(__builtin_amdgcn_global_load_async_to_lds_b128)
#define HAVE_ASYNC 1
#else
#define HAVE_ASYNC 0
#endif

__device__ __forceinline__ void cp16(const unsigned short* g, unsigned short* l){
#if HAVE_ASYNC
  __builtin_amdgcn_global_load_async_to_lds_b128(
      (__attribute__((address_space(1))) v4i*)(uintptr_t)g,
      (__attribute__((address_space(3))) v4i*)(unsigned int)(uintptr_t)l, 0, 0);
#else
  *(uint4*)l = *(const uint4*)g;
#endif
}

#if HAVE_ASYNC
#  if __has_builtin(__builtin_amdgcn_s_wait_asynccnt)
#    define WAIT_ASYNC(n) __builtin_amdgcn_s_wait_asynccnt(n)
#  else
#    define WAIT_ASYNC(n) asm volatile("s_wait_asynccnt %0" :: "i"(n) : "memory")
#  endif
#else
#  define WAIT_ASYNC(n) ((void)0)
#endif

// Load one 16x16x32 fragment from an LDS tile laid out row-major (ushort, given stride).
// "row" must already include (lane & 15). K packing per ISA 7.12.2:
// lanes 0-15 take K {0..7,16..23}+kbase, lanes 16-31 take K {8..15,24..31}+kbase.
__device__ __forceinline__ v16bf lds_frag(const unsigned short* tile, int stride,
                                          int row, int kbase, int lane){
  FragBF r;
  const unsigned short* p = tile + row*stride + kbase + ((lane >> 4) << 3);
  #pragma unroll
  for (int vv = 0; vv < 8; ++vv){
    int klo = (vv < 4) ? (2*vv) : (2*vv + 8);
    r.u[vv] = *(const unsigned int*)(p + klo);
  }
  return r.v;
}

// ---------------------------------------------------------------------------
// GEMM: C = act(A @ W + bias). A: bf16 (M,K) row-major. BT: bf16 W^T (N,K)
// row-major. Outputs: optional f32 C and/or bf16 Cb. Tile 128x64, BK=64,
// double-buffered async LDS staging, 8 waves (16x64 strip each).
// ---------------------------------------------------------------------------
#define GBM 128
#define GBN 64
#define GBK 64
#define GSTR 72
#define ASZ (GBM*GSTR)
#define BSZ (GBN*GSTR)

__device__ __forceinline__ void gemm_stage(const unsigned short* __restrict__ A,
                                           const unsigned short* __restrict__ BT,
                                           unsigned short* sAb, unsigned short* sBb,
                                           int m0, int n0, int k0, int K, int tid){
  #pragma unroll
  for (int i = 0; i < 4; ++i){               // A tile: 128 rows x 8 chunks
    int c = tid + 256*i, row = c >> 3, kc = c & 7;
    cp16(A + (size_t)(m0 + row)*K + k0 + kc*8, sAb + row*GSTR + kc*8);
  }
  #pragma unroll
  for (int i = 0; i < 2; ++i){               // B tile: 64 rows x 8 chunks
    int c = tid + 256*i, row = c >> 3, kc = c & 7;
    cp16(BT + (size_t)(n0 + row)*K + k0 + kc*8, sBb + row*GSTR + kc*8);
  }
}

__global__ __launch_bounds__(256)
void gemm_kernel(const unsigned short* __restrict__ A, const unsigned short* __restrict__ BT,
                 const float* __restrict__ bias, float* __restrict__ C,
                 unsigned short* __restrict__ Cb, int M, int K, int N, int act){
  __shared__ __align__(16) unsigned short sA[2*ASZ];
  __shared__ __align__(16) unsigned short sB[2*BSZ];
  const int tid  = threadIdx.x;
  const int lane = tid & 31;
  const int wave = tid >> 5;
  const int hf   = lane >> 4;
  const int l16  = lane & 15;
  const int m0 = blockIdx.y * GBM;
  const int n0 = blockIdx.x * GBN;

  v8f acc[4];
  #pragma unroll
  for (int i = 0; i < 4; ++i){ v8f z = {0,0,0,0,0,0,0,0}; acc[i] = z; }

  const int nst = K / GBK;
  gemm_stage(A, BT, sA, sB, m0, n0, 0, K, tid);
  int buf = 0;
  for (int s = 0; s < nst; ++s){
    if (s + 1 < nst){
      gemm_stage(A, BT, sA + (buf^1)*ASZ, sB + (buf^1)*BSZ, m0, n0, (s+1)*GBK, K, tid);
      WAIT_ASYNC(6);                         // current stage's 6 asyncs retired
    } else {
      WAIT_ASYNC(0);
    }
    __syncthreads();
    const unsigned short* sAb = sA + buf*ASZ;
    const unsigned short* sBb = sB + buf*BSZ;
    #pragma unroll
    for (int kk = 0; kk < GBK; kk += 32){
      v16bf a = lds_frag(sAb, GSTR, wave*16 + l16, kk, lane);
      #pragma unroll
      for (int nt = 0; nt < 4; ++nt){
        v16bf b = lds_frag(sBb, GSTR, nt*16 + l16, kk, lane);
        acc[nt] = wmma_bf16(a, b, acc[nt]);
      }
    }
    __syncthreads();                         // compute(buf) done before restaging it
    buf ^= 1;
  }
  #pragma unroll
  for (int nt = 0; nt < 4; ++nt){
    #pragma unroll
    for (int r = 0; r < 8; ++r){
      int grow = m0 + wave*16 + r + 8*hf;
      int gcol = n0 + nt*16 + l16;
      float val = acc[nt][r] + bias[gcol];
      if (act) val = fmaxf(val, 0.f);
      if (C)  C[(size_t)grow*N + gcol] = val;
      if (Cb) Cb[(size_t)grow*N + gcol] = f2bf(val);
    }
  }
}

// ---------------------------------------------------------------------------
// Fused flash attention per (b, h, 128 q-rows). Q/K/V: bf16 (bs*T, 512),
// head h = columns [h*64, h*64+64). Online softmax, pair + causal masks.
// ---------------------------------------------------------------------------
#define AQT 128
#define AKT 64
#define ASTR 72

__global__ __launch_bounds__(256)
void attn_kernel(const unsigned short* __restrict__ Q, const unsigned short* __restrict__ K,
                 const unsigned short* __restrict__ V, const float* __restrict__ qmask,
                 const float* __restrict__ kmask, float* __restrict__ O, int causal){
  __shared__ __align__(16) unsigned short sK[AKT*ASTR];     // [tk][feat]
  __shared__ __align__(16) unsigned short sV[DHEAD*ASTR];   // [feat][tk] (transposed)
  __shared__ __align__(16) unsigned short sP[8*16*ASTR];    // per-wave P tiles
  __shared__ float qm_s[AQT];
  __shared__ float km_s[AKT];

  const int tid = threadIdx.x, lane = tid & 31, wave = tid >> 5;
  const int hf = lane >> 4, l16 = lane & 15;
  const int qt = blockIdx.x, h = blockIdx.y, b = blockIdx.z;
  const int q0 = qt * AQT;
  const int qrow = q0 + wave*16 + l16;
  const float scale = 0.04419417382415922f;                 // 1/sqrt(512)

  FragBF aq[2];                                             // Q fragments (dh=64)
  {
    const unsigned short* qbase = Q + (size_t)(b*SEQ_T + qrow)*D_MODEL + h*DHEAD + hf*8;
    #pragma unroll
    for (int kk2 = 0; kk2 < 2; ++kk2){
      #pragma unroll
      for (int vv = 0; vv < 8; ++vv){
        int klo = ((vv < 4) ? (2*vv) : (2*vv + 8)) + kk2*32;
        aq[kk2].u[vv] = *(const unsigned int*)(qbase + klo);
      }
    }
  }
  if (tid < AQT) qm_s[tid] = qmask[b*SEQ_T + q0 + tid];
  __syncthreads();

  float mrow[8], lrow[8];
  v8f oacc[4];
  #pragma unroll
  for (int r = 0; r < 8; ++r){ mrow[r] = -INFINITY; lrow[r] = 0.f; }
  #pragma unroll
  for (int i = 0; i < 4; ++i){ v8f z = {0,0,0,0,0,0,0,0}; oacc[i] = z; }

  const int kbmax = causal ? (q0 + AQT) : SEQ_T;
  unsigned short* pw = sP + wave*16*ASTR;

  for (int kb = 0; kb < kbmax; kb += AKT){
    __syncthreads();
    #pragma unroll
    for (int i = 0; i < 2; ++i){                            // K tile: async 16B chunks
      int c = tid + 256*i, tk = c >> 3, kc = c & 7;
      cp16(K + (size_t)(b*SEQ_T + kb + tk)*D_MODEL + h*DHEAD + kc*8,
           sK + tk*ASTR + kc*8);
    }
    #pragma unroll
    for (int i = 0; i < 16; ++i){                           // V tile: manual transpose
      int idx = i*256 + tid, tk = idx >> 6, ff = idx & 63;
      sV[ff*ASTR + tk] = V[(size_t)(b*SEQ_T + kb + tk)*D_MODEL + h*DHEAD + ff];
    }
    if (tid < AKT) km_s[tid] = kmask[b*SEQ_T + kb + tid];
    WAIT_ASYNC(0);
    __syncthreads();

    v8f s[4];                                               // S = Q K^T
    #pragma unroll
    for (int ct = 0; ct < 4; ++ct){
      v8f z = {0,0,0,0,0,0,0,0};
      v16bf b0 = lds_frag(sK, ASTR, ct*16 + l16, 0, lane);
      z = wmma_bf16(aq[0].v, b0, z);
      v16bf b1 = lds_frag(sK, ASTR, ct*16 + l16, 32, lane);
      z = wmma_bf16(aq[1].v, b1, z);
      s[ct] = z;
    }
    #pragma unroll
    for (int ct = 0; ct < 4; ++ct){                         // scale + masks
      int tk = kb + ct*16 + l16;
      float km = km_s[ct*16 + l16];
      #pragma unroll
      for (int r = 0; r < 8; ++r){
        int qg = q0 + wave*16 + r + 8*hf;
        float val = s[ct][r] * scale;
        float qm = qm_s[wave*16 + r + 8*hf];
        if (qm * km <= 0.f) val = NEG_BIG;
        if (causal && tk > qg) val = NEG_BIG;
        s[ct][r] = val;
      }
    }
    #pragma unroll
    for (int r = 0; r < 8; ++r){                            // online softmax
      float mx = fmaxf(fmaxf(s[0][r], s[1][r]), fmaxf(s[2][r], s[3][r]));
      #pragma unroll
      for (int off = 1; off < 16; off <<= 1)
        mx = fmaxf(mx, __shfl_xor(mx, off, 32));
      float mnew = fmaxf(mrow[r], mx);
      float alpha = __expf(mrow[r] - mnew);
      float rs = 0.f;
      #pragma unroll
      for (int ct = 0; ct < 4; ++ct){
        float p = __expf(s[ct][r] - mnew);
        s[ct][r] = p;
        rs += p;
      }
      #pragma unroll
      for (int off = 1; off < 16; off <<= 1)
        rs += __shfl_xor(rs, off, 32);
      lrow[r] = lrow[r]*alpha + rs;
      mrow[r] = mnew;
      #pragma unroll
      for (int ct2 = 0; ct2 < 4; ++ct2) oacc[ct2][r] *= alpha;
    }
    #pragma unroll
    for (int ct = 0; ct < 4; ++ct)                          // spill P (bf16)
      #pragma unroll
      for (int r = 0; r < 8; ++r)
        pw[(r + 8*hf)*ASTR + ct*16 + l16] = f2bf(s[ct][r]);
    __syncthreads();
    #pragma unroll
    for (int kk2 = 0; kk2 < 2; ++kk2){                      // O += P @ V
      v16bf ap = lds_frag(pw, ASTR, l16, kk2*32, lane);
      #pragma unroll
      for (int ct2 = 0; ct2 < 4; ++ct2){
        v16bf bv = lds_frag(sV, ASTR, ct2*16 + l16, kk2*32, lane);
        oacc[ct2] = wmma_bf16(ap, bv, oacc[ct2]);
      }
    }
  }
  #pragma unroll
  for (int ct2 = 0; ct2 < 4; ++ct2){
    #pragma unroll
    for (int r = 0; r < 8; ++r){
      int qg = q0 + wave*16 + r + 8*hf;
      O[(size_t)(b*SEQ_T + qg)*D_MODEL + h*DHEAD + ct2*16 + l16] = oacc[ct2][r] / lrow[r];
    }
  }
}

// ---------------------------------------------------------------------------
// y = LayerNorm(x * rowmask + resid); writes f32 out and bf16 outb.
// ---------------------------------------------------------------------------
__global__ __launch_bounds__(256)
void resln_kernel(const float* __restrict__ x, const float* __restrict__ resid,
                  const float* __restrict__ mask, const float* __restrict__ g,
                  const float* __restrict__ bb, float* __restrict__ out,
                  unsigned short* __restrict__ outb){
  __shared__ float2 red[256];
  const int row = blockIdx.x, tid = threadIdx.x;
  const float m = mask ? mask[row] : 1.0f;
  const float2 xv = ((const float2*)(x + (size_t)row*D_MODEL))[tid];
  const float2 rv = ((const float2*)(resid + (size_t)row*D_MODEL))[tid];
  float a0 = xv.x*m + rv.x;
  float a1 = xv.y*m + rv.y;
  red[tid] = make_float2(a0 + a1, a0*a0 + a1*a1);
  __syncthreads();
  for (int s = 128; s > 0; s >>= 1){
    if (tid < s){ float2 o = red[tid+s]; red[tid].x += o.x; red[tid].y += o.y; }
    __syncthreads();
  }
  const float mean = red[0].x * (1.0f/D_MODEL);
  const float var  = red[0].y * (1.0f/D_MODEL) - mean*mean;
  const float inv  = rsqrtf(var + 1e-5f);
  float o0 = (a0 - mean)*inv*g[2*tid]   + bb[2*tid];
  float o1 = (a1 - mean)*inv*g[2*tid+1] + bb[2*tid+1];
  ((float2*)(out + (size_t)row*D_MODEL))[tid] = make_float2(o0, o1);
  *(unsigned int*)(outb + (size_t)row*D_MODEL + 2*tid) = packbf(o0, o1);
}

// out[row] = (word[tok[row]] + pos[pidx[row]]) * mask[row]; dual f32/bf16 write
__global__ __launch_bounds__(256)
void embed_kernel(const float* __restrict__ word, const float* __restrict__ pos,
                  const int* __restrict__ tok, const int* __restrict__ pidx,
                  const float* __restrict__ mask, float* __restrict__ out,
                  unsigned short* __restrict__ outb){
  const int row = blockIdx.x, tid = threadIdx.x;
  const int t = tok[row], p = pidx[row];
  const float m = mask[row];
  const float2 w  = ((const float2*)(word + (size_t)t*D_MODEL))[tid];
  const float2 pp = ((const float2*)(pos  + (size_t)p*D_MODEL))[tid];
  float o0 = (w.x + pp.x)*m, o1 = (w.y + pp.y)*m;
  ((float2*)(out + (size_t)row*D_MODEL))[tid] = make_float2(o0, o1);
  *(unsigned int*)(outb + (size_t)row*D_MODEL + 2*tid) = packbf(o0, o1);
}

// WT[n][k] = bf16(W[k][n]) : 32x32 LDS tile transpose, block (32,8)
__global__ __launch_bounds__(256)
void wt_kernel(const float* __restrict__ W, unsigned short* __restrict__ WT, int K, int N){
  __shared__ float tile[32][33];
  const int k0 = blockIdx.y*32, n0 = blockIdx.x*32;
  const int tx = threadIdx.x, ty = threadIdx.y;
  #pragma unroll
  for (int i = 0; i < 32; i += 8)
    tile[ty+i][tx] = W[(size_t)(k0+ty+i)*N + (n0+tx)];
  __syncthreads();
  #pragma unroll
  for (int i = 0; i < 32; i += 8)
    WT[(size_t)(n0+ty+i)*K + (k0+tx)] = f2bf(tile[tx][ty+i]);
}

// ---------------------------------------------------------------------------
// Host orchestration
// ---------------------------------------------------------------------------
struct AttnP {
  const unsigned short *WqT, *WkT, *WvT;
  const float *bq, *bk, *bv, *lng, *lnb;
};

static void gemm(hipStream_t s, const unsigned short* A, const unsigned short* BT,
                 const float* bias, float* C, unsigned short* Cb, int M, int K, int N, int act){
  dim3 grid(N / GBN, M / GBM);
  gemm_kernel<<<grid, 256, 0, s>>>(A, BT, bias, C, Cb, M, K, N, act);
}

static void mha(hipStream_t s, const float* qin_f, const unsigned short* qin_b,
                const float* qm, const unsigned short* kin_b, const float* km,
                const AttnP& p, int causal, float* out_f, unsigned short* out_b,
                unsigned short* qbb, unsigned short* kbb, unsigned short* vbb, float* atto){
  gemm(s, qin_b, p.WqT, p.bq, nullptr, qbb, NROWS, D_MODEL, D_MODEL, 1);
  gemm(s, kin_b, p.WkT, p.bk, nullptr, kbb, NROWS, D_MODEL, D_MODEL, 1);
  gemm(s, kin_b, p.WvT, p.bv, nullptr, vbb, NROWS, D_MODEL, D_MODEL, 1);
  attn_kernel<<<dim3(SEQ_T/AQT, NHEAD, BATCH), 256, 0, s>>>(qbb, kbb, vbb, qm, km, atto, causal);
  resln_kernel<<<NROWS, 256, 0, s>>>(atto, qin_f, qm, p.lng, p.lnb, out_f, out_b);
}

extern "C" void kernel_launch(void* const* d_in, const int* in_sizes, int n_in,
                              void* d_out, int out_size, void* d_ws, size_t ws_size,
                              hipStream_t stream) {
  (void)in_sizes; (void)n_in; (void)out_size; (void)ws_size;
  auto F = [&](int i){ return (const float*)d_in[i]; };

  // Leaf order: jax tree (dict keys sorted).
  // 0 cls_W, 1 cls_b; dec layers base 2+22*i; enc layers base 134+14*i;
  // 218 src_pos, 219 src_word, 220 trg_pos, 221 trg_word;
  // 222 src, 223 src_mask, 224 src_position, 225 trg, 226 trg_mask, 227 trg_position
  const int*   src      = (const int*)d_in[222];
  const float* src_mask = F(223);
  const int*   src_pidx = (const int*)d_in[224];
  const int*   trg      = (const int*)d_in[225];
  const float* trg_mask = F(226);
  const int*   trg_pidx = (const int*)d_in[227];

  float* ws = (float*)d_ws;
  const size_t S = (size_t)NROWS * D_MODEL;   // 2M elements / slot
  // f32 slots
  float* enc_x = ws + 0*S;
  float* dec_x = ws + 1*S;
  float* atto  = ws + 2*S;
  float* t1    = ws + 3*S;
  float* t2    = ws + 4*S;
  float* ffo   = ws + 5*S;
  // bf16 region
  unsigned short* wsb   = (unsigned short*)(ws + 6*S);
  unsigned short* enc_xb = wsb + 0*S;
  unsigned short* dec_xb = wsb + 1*S;
  unsigned short* qbb    = wsb + 2*S;
  unsigned short* kbb    = wsb + 3*S;
  unsigned short* vbb    = wsb + 4*S;
  unsigned short* t1b    = wsb + 5*S;
  unsigned short* t2b    = wsb + 6*S;
  unsigned short* ffhb   = wsb + 7*S;          // 4 slots (4096 x 2048)
  unsigned short* wcur   = wsb + 11*S;         // transposed bf16 weights

  auto make_wt = [&](const float* W, int K, int N)->unsigned short*{
    unsigned short* p = wcur; wcur += (size_t)K*N;
    wt_kernel<<<dim3(N/32, K/32), dim3(32,8), 0, stream>>>(W, p, K, N);
    return p;
  };
  auto attn_wt = [&](int base)->AttnP{          // sorted: Wk,Wq,Wv,bk,bq,bv,ln_b,ln_g
    AttnP a;
    a.WkT = make_wt(F(base+0), D_MODEL, D_MODEL);
    a.WqT = make_wt(F(base+1), D_MODEL, D_MODEL);
    a.WvT = make_wt(F(base+2), D_MODEL, D_MODEL);
    a.bk = F(base+3); a.bq = F(base+4); a.bv = F(base+5);
    a.lnb = F(base+6); a.lng = F(base+7);
    return a;
  };

  // ---- pre-transpose/convert all GEMM weights to bf16 W^T ----
  const unsigned short* encW1T[6]; const unsigned short* encW2T[6]; AttnP encSelf[6];
  const unsigned short* decW1T[6]; const unsigned short* decW2T[6]; AttnP decSelf[6], decCross[6];
  for (int L = 0; L < 6; ++L){
    int base = 134 + L*14;                      // enc: W1,W2,b1,b2,ln_b,ln_g,self(8)
    encW1T[L] = make_wt(F(base+0), D_MODEL, FF_DIM);
    encW2T[L] = make_wt(F(base+1), FF_DIM, D_MODEL);
    encSelf[L] = attn_wt(base+6);
  }
  for (int L = 0; L < 6; ++L){
    int base = 2 + L*22;                        // dec: W1,W2,b1,b2,cross(8),ln_b,ln_g,self(8)
    decW1T[L] = make_wt(F(base+0), D_MODEL, FF_DIM);
    decW2T[L] = make_wt(F(base+1), FF_DIM, D_MODEL);
    decCross[L] = attn_wt(base+4);
    decSelf[L]  = attn_wt(base+14);
  }
  const unsigned short* clsWT = make_wt(F(0), D_MODEL, VOCAB);
  const float* cls_b = F(1);

  // ---- embeddings ----
  embed_kernel<<<NROWS, 256, 0, stream>>>(F(219), F(218), src, src_pidx, src_mask, enc_x, enc_xb);
  embed_kernel<<<NROWS, 256, 0, stream>>>(F(221), F(220), trg, trg_pidx, trg_mask, dec_x, dec_xb);

  // ---- encoder ----
  for (int L = 0; L < 6; ++L){
    int base = 134 + L*14;
    const float *b1 = F(base+2), *b2 = F(base+3), *lnb = F(base+4), *lng = F(base+5);
    mha(stream, enc_x, enc_xb, src_mask, enc_xb, src_mask, encSelf[L], 0,
        t1, t1b, qbb, kbb, vbb, atto);
    gemm(stream, t1b,  encW1T[L], b1, nullptr, ffhb, NROWS, D_MODEL, FF_DIM, 1);
    gemm(stream, ffhb, encW2T[L], b2, ffo, nullptr, NROWS, FF_DIM, D_MODEL, 0);
    resln_kernel<<<NROWS, 256, 0, stream>>>(ffo, enc_x, nullptr, lng, lnb, enc_x, enc_xb);
  }

  // ---- decoder ----
  for (int L = 0; L < 6; ++L){
    int base = 2 + L*22;
    const float *b1 = F(base+2), *b2 = F(base+3), *lnb = F(base+12), *lng = F(base+13);
    mha(stream, dec_x, dec_xb, trg_mask, dec_xb, trg_mask, decSelf[L], 1,
        t1, t1b, qbb, kbb, vbb, atto);
    mha(stream, t1, t1b, trg_mask, enc_xb, src_mask, decCross[L], 0,
        t2, t2b, qbb, kbb, vbb, atto);
    gemm(stream, t2b,  decW1T[L], b1, nullptr, ffhb, NROWS, D_MODEL, FF_DIM, 1);
    gemm(stream, ffhb, decW2T[L], b2, ffo, nullptr, NROWS, FF_DIM, D_MODEL, 0);
    resln_kernel<<<NROWS, 256, 0, stream>>>(ffo, dec_x, nullptr, lng, lnb, dec_x, dec_xb);
  }

  // ---- classifier ----
  gemm(stream, dec_xb, clsWT, cls_b, (float*)d_out, nullptr, NROWS, D_MODEL, VOCAB, 0);
}
